// Yolov2Recall_85152021610722
// MI455X (gfx1250) — compile-verified, matrix operand order checked by reading, not run on hardware
//
#include <hip/hip_runtime.h>
#include <hip/hip_bf16.h>
#include <math.h>

#define BATCH   8
#define NANCH   5
#define NCLS    20
#define GH      19
#define GW      19
#define NBOX    (NANCH*GH*GW)   // 1805
#define NSORT   2048
#define NGT     50
#define NTHR    1024
#define CONF_T  0.5f
#define NMS_T   0.45f
#define IOU_T   0.5f

typedef __attribute__((ext_vector_type(8))) int v8i;

__constant__ float c_anchors[2*NANCH] = {
    1.3221f, 1.73145f, 3.19275f, 4.00944f, 5.05587f, 8.09892f,
    9.47112f, 4.84053f, 11.2364f, 10.0071f };

__device__ __forceinline__ float sigmoidf_(float x) { return 1.f / (1.f + expf(-x)); }

__device__ __forceinline__ float iou_cxcywh(float4 a, float4 b) {
    float ax1 = a.x - a.z * 0.5f, ax2 = a.x + a.z * 0.5f;
    float ay1 = a.y - a.w * 0.5f, ay2 = a.y + a.w * 0.5f;
    float bx1 = b.x - b.z * 0.5f, bx2 = b.x + b.z * 0.5f;
    float by1 = b.y - b.w * 0.5f, by2 = b.y + b.w * 0.5f;
    float uw = fmaxf(ax2, bx2) - fminf(ax1, bx1);
    float uh = fmaxf(ay2, by2) - fminf(ay1, by1);
    float cw = a.z + b.z - uw;
    float ch = a.w + b.w - uh;
    float carea = (cw > 0.f && ch > 0.f) ? cw * ch : 0.f;
    float uarea = a.z * a.w + b.z * b.w - carea;
    return carea / uarea;
}

// ---- Kernel 0: zero the scalar accumulators (ws is poisoned by harness) ----
__global__ void k_init(int* counters) {
    if (threadIdx.x < 3) counters[threadIdx.x] = 0;
}

// ---- Kernel 1: anchor decode -> boxes (cx,cy,w,h) + det_conf in workspace ----
__global__ void k_decode(const float* __restrict__ out,
                         float4* __restrict__ boxes, float* __restrict__ conf) {
    int idx = blockIdx.x * blockDim.x + threadIdx.x;
    if (idx >= BATCH * NBOX) return;
    int b = idx / NBOX, n = idx % NBOX;
    int a = n / (GH * GW), r = n % (GH * GW);
    int gy = r / GW, gx = r % GW;
    const float* p = out + ((size_t)(b * NANCH + a) * (5 + NCLS)) * (GH * GW) + r;
    float o0 = p[0];
    float o1 = p[1 * GH * GW];
    float o2 = p[2 * GH * GW];
    float o3 = p[3 * GH * GW];
    float o4 = p[4 * GH * GW];
    float cx = (sigmoidf_(o0) + (float)gx) * (1.f / (float)GW);
    float cy = (sigmoidf_(o1) + (float)gy) * (1.f / (float)GH);
    float bw = expf(o2) * (c_anchors[2 * a]     / (float)GW);
    float bh = expf(o3) * (c_anchors[2 * a + 1] / (float)GH);
    boxes[idx] = make_float4(cx, cy, bw, bh);
    conf[idx]  = sigmoidf_(o4);
}

// ---- Kernel 2: per-batch sort + greedy NMS + GT matching, all in LDS ----
__global__ __launch_bounds__(NTHR)
void k_batch(const float4* __restrict__ boxes, const float* __restrict__ conf,
             const float* __restrict__ target, int* __restrict__ counters) {
    __shared__ float  skey[NSORT];
    __shared__ int    sidx[NSORT];
    __shared__ float4 sbox[NSORT];
    __shared__ int    skeep[NSORT];
    __shared__ float  struth[NGT * 5];
    __shared__ int    svalid[NGT];

    const int b   = blockIdx.x;
    const int tid = threadIdx.x;
    const float*  cb = conf  + (size_t)b * NBOX;
    const float4* bb = boxes + (size_t)b * NBOX;

    // load conf keys; pad with sentinel below any possible sigmoid output
    for (int i = tid; i < NSORT; i += NTHR) {
        skey[i] = (i < NBOX) ? cb[i] : -1.0f;
        sidx[i] = i;
    }
    __syncthreads();

    // bitonic sort: conf descending, index ascending (matches stable argsort(-det))
    for (int k = 2; k <= NSORT; k <<= 1) {
        for (int j = k >> 1; j > 0; j >>= 1) {
            for (int it = 0; it < NSORT / NTHR; ++it) {
                int i = tid + it * NTHR;
                int ixj = i ^ j;
                if (ixj > i) {
                    float ka = skey[i], kb = skey[ixj];
                    int   ia = sidx[i], ib = sidx[ixj];
                    bool up  = ((i & k) == 0);
                    // true if element at ixj should come before element at i
                    bool mis = (kb > ka) || (kb == ka && ib < ia);
                    if (mis == up) {
                        skey[i] = kb; skey[ixj] = ka;
                        sidx[i] = ib; sidx[ixj] = ia;
                    }
                }
            }
            __syncthreads();
        }
    }

    // gather sorted boxes into LDS via CDNA5 async global->LDS DMA (ASYNCcnt path)
    for (int it = 0; it < NSORT / NTHR; ++it) {
        int jj = tid + it * NTHR;
        if (jj < NBOX) {
            const float4* gp = bb + sidx[jj];
            unsigned long long ga = (unsigned long long)(uintptr_t)gp;
            unsigned int       la = (unsigned int)(uintptr_t)(&sbox[jj]);
            asm volatile("global_load_async_to_lds_b128 %0, %1, off"
                         :: "v"(la), "v"(ga) : "memory");
        }
        if (jj < NSORT) skeep[jj] = (jj < NBOX) && (skey[jj] > CONF_T);
    }
    asm volatile("s_wait_asynccnt 0" ::: "memory");
    __syncthreads();

    // sequential greedy NMS over conf-sorted boxes
    for (int i = 0; i < NBOX; ++i) {
        __syncthreads();
        if (!skeep[i]) continue;        // uniform across block
        float4 bi = sbox[i];
        for (int it = 0; it < NSORT / NTHR; ++it) {
            int jj = tid + it * NTHR;
            if (jj > i && jj < NBOX && skeep[jj]) {
                if (iou_cxcywh(bi, sbox[jj]) > NMS_T) skeep[jj] = 0;
            }
        }
    }
    __syncthreads();

    // proposals = sum(keep): popcount of 2048 0/1 flags on the matrix pipe.
    // B = all-ones bytes -> D[m][n] = rowsum(A[m]); K-permutation invariant, so
    // any bijective packing of flags into the A operand gives exact row sums.
    // Two chained V_WMMA_I32_16X16X64_IU8 accumulate all 2048 flags.
    if (tid < 32) {                     // wave 0 only, full EXEC within the wave
        v8i a0, a1, bones, acc;
        for (int e = 0; e < 8; ++e) {
            int base = tid * 32 + e * 4;
            int p0 = 0, p1 = 0;
            for (int c = 0; c < 4; ++c) {
                p0 |= (skeep[base + c]        & 1) << (8 * c);
                p1 |= (skeep[1024 + base + c] & 1) << (8 * c);
            }
            a0[e] = p0; a1[e] = p1;
            bones[e] = 0x01010101;
            acc[e] = 0;
        }
        acc = __builtin_amdgcn_wmma_i32_16x16x64_iu8(false, a0, false, bones, acc, false, false);
        acc = __builtin_amdgcn_wmma_i32_16x16x64_iu8(false, a1, false, bones, acc, false, false);
        // lane<16: acc[r]=rowsum[M=r]; lane>=16: acc[r]=rowsum[M=8+r] (col N=lane%16)
        int s = acc[0] + acc[1] + acc[2] + acc[3] + acc[4] + acc[5] + acc[6] + acc[7];
        s += __shfl_xor(s, 16, 32);     // lane0: sum over M=0..15 = total proposals
        if (tid == 0) atomicAdd(&counters[1], s);
    }
    __syncthreads();

    // ground truth: rows [cls,cx,cy,w,h]; valid while cumulative cx != 0
    for (int t = tid; t < NGT * 5; t += NTHR) struth[t] = target[(size_t)b * NGT * 5 + t];
    __syncthreads();
    if (tid == 0) {
        bool run = true; int tot = 0;
        for (int t = 0; t < NGT; ++t) {
            run = run && (struth[t * 5 + 1] != 0.f);
            svalid[t] = run ? 1 : 0;
            tot += run ? 1 : 0;
        }
        atomicAdd(&counters[0], tot);
    }
    __syncthreads();

    // one wave32 per GT: max IoU against kept boxes, count hits > IOU_T
    int wave = tid >> 5, lane = tid & 31;
    for (int t = wave; t < NGT; t += (NTHR >> 5)) {
        if (!svalid[t]) continue;       // uniform within wave
        float4 g = make_float4(struth[t * 5 + 1], struth[t * 5 + 2],
                               struth[t * 5 + 3], struth[t * 5 + 4]);
        float best = 0.f;
        for (int j = lane; j < NBOX; j += 32)
            if (skeep[j]) best = fmaxf(best, iou_cxcywh(g, sbox[j]));
        for (int off = 16; off > 0; off >>= 1)
            best = fmaxf(best, __shfl_xor(best, off, 32));
        if (lane == 0 && best > IOU_T) atomicAdd(&counters[2], 1);
    }
}

// ---- Kernel 3: scalars -> [total, proposals, correct, precision, recall, fscore] ----
__global__ void k_final(const int* __restrict__ counters, float* __restrict__ out6) {
    if (threadIdx.x == 0 && blockIdx.x == 0) {
        float tf = (float)counters[0];
        float pf = (float)counters[1];
        float cf = (float)counters[2];
        float prec = cf / (pf + 1e-6f);
        float rec  = cf / (tf + 1e-6f);
        float f    = 2.f * prec * rec / (prec + rec + 1e-6f);
        out6[0] = tf; out6[1] = pf; out6[2] = cf;
        out6[3] = prec; out6[4] = rec; out6[5] = f;
    }
}

extern "C" void kernel_launch(void* const* d_in, const int* in_sizes, int n_in,
                              void* d_out, int out_size, void* d_ws, size_t ws_size,
                              hipStream_t stream) {
    (void)in_sizes; (void)n_in; (void)out_size; (void)ws_size;
    const float* out_t  = (const float*)d_in[0];   // [8,125,19,19]
    const float* target = (const float*)d_in[1];   // [8,250]

    int*    counters = (int*)d_ws;                                       // 3 ints
    float4* boxes    = (float4*)((char*)d_ws + 256);                     // B*N float4
    float*  conf     = (float*)((char*)d_ws + 256 +
                                 sizeof(float4) * (size_t)BATCH * NBOX); // B*N float

    k_init  <<<1, 64, 0, stream>>>(counters);
    k_decode<<<(BATCH * NBOX + 255) / 256, 256, 0, stream>>>(out_t, boxes, conf);
    k_batch <<<BATCH, NTHR, 0, stream>>>(boxes, conf, target, counters);
    k_final <<<1, 1, 0, stream>>>(counters, (float*)d_out);
}
